// PaiNN_19146964206355
// MI455X (gfx1250) — matrix operand order repaired and checked by reading, not compiled, not run on hardware
//
#include <hip/hip_runtime.h>
#include <hip/hip_bf16.h>
#include <math.h>

#define N_NODES 50000
#define N_EDGES 800000
#define HDIM 128
#define RDIM 20
#define LAYERS 3
#define CUTOFF_F 5.0f
#define PI_F 3.14159265358979f

typedef __attribute__((ext_vector_type(16))) __bf16 v16bf;
typedef __attribute__((ext_vector_type(8)))  __bf16 v8bf;
typedef __attribute__((ext_vector_type(8)))  float  v8f;

__device__ __forceinline__ float silu_f(float x) {
    return x * (1.0f / (1.0f + __expf(-x)));
}

__device__ __forceinline__ void atomAddF(float* p, float val) {
    unsafeAtomicAdd(p, val);
}

// ---------------------------------------------------------------------------
// LDS-tiled GEMM: C(M,N) = act(A(M,K) @ B(K,N) + bias)
// fp32 in global, bf16 WMMA math, fp32 accumulate.
// K is compile-time (128 or 256), Ncols % 64 == 0.
// Block: 256 threads (8 waves) -> 64x64 output tile.
// Wave w: rows (w&3)*16..+15, cols (w>>2)*32..+31 (two 16x16 WMMA tiles).
// LDS: As[64][40] bf16 row-major, Bs[64][40] bf16 transposed (Bs[col][k]).
// Padded stride 40 -> fragment ds_load_b128 hits 16 distinct banks.
// ---------------------------------------------------------------------------
template<int K>
__global__ void gemm_tiled(const float* __restrict__ A, const float* __restrict__ B,
                           const float* __restrict__ bias, float* __restrict__ C,
                           int M, int Ncols, int act)
{
    constexpr int LDA = 40;
    __shared__ __bf16 As[64 * LDA];
    __shared__ __bf16 Bs[64 * LDA];

    int tid = threadIdx.x;
    int nTilesN = Ncols >> 6;
    int bM = blockIdx.x / nTilesN;
    int bN = blockIdx.x % nTilesN;
    int rowBase = bM * 64;
    int colBase = bN * 64;

    int w    = tid >> 5;
    int lane = tid & 31;
    int half = lane >> 4;
    int lr   = lane & 15;

    int wr = (w & 3) * 16;   // wave row offset within block tile
    int wc = (w >> 2) * 32;  // wave col offset within block tile (2 tiles)

    // staging thread mapping
    int arow = tid >> 2;          // 0..63
    int aseg = (tid & 3) * 8;     // 0,8,16,24
    int bk   = tid >> 3;          // 0..31
    int bcg  = (tid & 7) * 8;     // 0..56

    v8f acc0 = {}, acc1 = {};

    #pragma unroll
    for (int k0 = 0; k0 < K; k0 += 32) {
        // ---- stage A tile (64 x 32) ----
        {
            long gr = (long)(rowBase + arow);
            float f[8];
            if (gr < (long)M) {
                const float4* p = reinterpret_cast<const float4*>(A + gr * K + k0 + aseg);
                float4 x = p[0], y = p[1];
                f[0]=x.x; f[1]=x.y; f[2]=x.z; f[3]=x.w;
                f[4]=y.x; f[5]=y.y; f[6]=y.z; f[7]=y.w;
            } else {
                #pragma unroll
                for (int i = 0; i < 8; i++) f[i] = 0.0f;
            }
            v8bf av;
            #pragma unroll
            for (int i = 0; i < 8; i++) av[i] = (__bf16)f[i];
            *reinterpret_cast<v8bf*>(&As[arow * LDA + aseg]) = av;
        }
        // ---- stage B tile (32 x 64), transposed into Bs[col][k] ----
        {
            const float4* p = reinterpret_cast<const float4*>(
                B + (long)(k0 + bk) * Ncols + colBase + bcg);
            float4 x = p[0], y = p[1];
            float f[8];
            f[0]=x.x; f[1]=x.y; f[2]=x.z; f[3]=x.w;
            f[4]=y.x; f[5]=y.y; f[6]=y.z; f[7]=y.w;
            #pragma unroll
            for (int j = 0; j < 8; j++)
                Bs[(bcg + j) * LDA + bk] = (__bf16)f[j];
            if (k0 + 32 < K)
                __builtin_prefetch(B + (long)(k0 + 32 + bk) * Ncols + colBase + bcg, 0, 0);
        }
        __syncthreads();

        // ---- fragments from LDS ----
        v16bf a;
        {
            v8bf lo = *reinterpret_cast<const v8bf*>(&As[(wr + lr) * LDA + 16 * half]);
            v8bf hi = *reinterpret_cast<const v8bf*>(&As[(wr + lr) * LDA + 16 * half + 8]);
            #pragma unroll
            for (int i = 0; i < 8; i++) { a[i] = lo[i]; a[8 + i] = hi[i]; }
        }
        v16bf b0, b1;
        {
            v8bf p0 = *reinterpret_cast<const v8bf*>(&Bs[(wc + lr) * LDA + 16 * half]);
            v8bf p1 = *reinterpret_cast<const v8bf*>(&Bs[(wc + lr) * LDA + 16 * half + 8]);
            v8bf q0 = *reinterpret_cast<const v8bf*>(&Bs[(wc + 16 + lr) * LDA + 16 * half]);
            v8bf q1 = *reinterpret_cast<const v8bf*>(&Bs[(wc + 16 + lr) * LDA + 16 * half + 8]);
            #pragma unroll
            for (int i = 0; i < 8; i++) { b0[i] = p0[i]; b0[8 + i] = p1[i]; }
            #pragma unroll
            for (int i = 0; i < 8; i++) { b1[i] = q0[i]; b1[8 + i] = q1[i]; }
        }
        acc0 = __builtin_amdgcn_wmma_f32_16x16x32_bf16(false, a, false, b0,
                                                       (short)0, acc0, false, false);
        acc1 = __builtin_amdgcn_wmma_f32_16x16x32_bf16(false, a, false, b1,
                                                       (short)0, acc1, false, false);
        __syncthreads();
    }

    // ---- epilogue: two 16x16 tiles ----
    int col0 = colBase + wc + lr;
    int col1 = col0 + 16;
    float bv0 = bias ? bias[col0] : 0.0f;
    float bv1 = bias ? bias[col1] : 0.0f;
    #pragma unroll
    for (int g = 0; g < 8; g++) {
        int row = rowBase + wr + g + 8 * half;
        if (row < M) {
            float x0 = acc0[g] + bv0;
            float x1 = acc1[g] + bv1;
            if (act) { x0 = silu_f(x0); x1 = silu_f(x1); }
            C[(long)row * Ncols + col0] = x0;
            C[(long)row * Ncols + col1] = x1;
        }
    }
}

// ---------------------------------------------------------------------------
// Per-edge message kernel. One wave per edge iteration.
// ---------------------------------------------------------------------------
__global__ void edge_kernel(const int* __restrict__ eidx, const float* __restrict__ edist,
                            const float* __restrict__ unit, const float* __restrict__ rbfw,
                            const float* __restrict__ rbfb, const float* __restrict__ mf,
                            const float* __restrict__ v, float* __restrict__ acc)
{
    __shared__ float sh_w[RDIM * 384];
    __shared__ float sh_b[384];
    for (int i = threadIdx.x; i < RDIM * 384; i += blockDim.x) sh_w[i] = rbfw[i];
    for (int i = threadIdx.x; i < 384; i += blockDim.x)        sh_b[i] = rbfb[i];
    __syncthreads();

    const float inv_sqrt3 = 0.57735026918962576f;
    int wave = threadIdx.x >> 5;
    int lane = threadIdx.x & 31;

    for (long e = (long)blockIdx.x * 8 + wave; e < N_EDGES; e += (long)gridDim.x * 8) {
        float dist = edist[e];
        float cosc = (dist <= CUTOFF_F) ? 0.5f * (__cosf(PI_F * dist / CUTOFF_F) + 1.0f) : 0.0f;
        float mine = 0.0f;
        if (lane < RDIM)
            mine = __sinf((float)(lane + 1) * PI_F * dist / CUTOFF_F) / dist * cosc;
        float rbf[RDIM];
        #pragma unroll
        for (int r = 0; r < RDIM; r++) rbf[r] = __shfl(mine, r, 32);

        int srcn = eidx[e];
        int dstn = eidx[N_EDGES + e];
        float u0 = unit[e * 3 + 0], u1 = unit[e * 3 + 1], u2 = unit[e * 3 + 2];
        const float* mfr = mf + (long)srcn * 384;
        const float* vr  = v  + (long)srcn * 384;
        float*       ar  = acc + (long)dstn * 512;

        #pragma unroll
        for (int kk = 0; kk < 4; kk++) {
            int c = lane + 32 * kk;
            float ws_ = sh_b[c], wvs = sh_b[128 + c], wvv = sh_b[256 + c];
            #pragma unroll
            for (int r = 0; r < RDIM; r++) {
                float rv = rbf[r];
                ws_ += rv * sh_w[r * 384 + c];
                wvs += rv * sh_w[r * 384 + 128 + c];
                wvv += rv * sh_w[r * 384 + 256 + c];
            }
            float ps  = mfr[c]       * ws_;
            float pvs = mfr[128 + c] * wvs;
            float pvv = mfr[256 + c] * wvv;
            atomAddF(&ar[c], ps);
            float c0 = vr[c]       * pvs * inv_sqrt3 + pvv * u0;
            float c1 = vr[128 + c] * pvs * inv_sqrt3 + pvv * u1;
            float c2 = vr[256 + c] * pvs * inv_sqrt3 + pvv * u2;
            atomAddF(&ar[128 + c], c0);
            atomAddF(&ar[256 + c], c1);
            atomAddF(&ar[384 + c], c2);
        }
    }
}

// ---------------------------------------------------------------------------
// Small elementwise kernels
// ---------------------------------------------------------------------------
__global__ void embed_kernel(const int* __restrict__ z, const float* __restrict__ emb,
                             float* __restrict__ s)
{
    long idx = (long)blockIdx.x * blockDim.x + threadIdx.x;
    if (idx < (long)N_NODES * HDIM) {
        int n = (int)(idx >> 7);
        int c = (int)(idx & 127);
        s[idx] = emb[(long)z[n] * HDIM + c];
    }
}

__global__ void unit_kernel(const float* __restrict__ vec, float* __restrict__ unit)
{
    int e = blockIdx.x * blockDim.x + threadIdx.x;
    if (e < N_EDGES) {
        float x = vec[3 * e], y = vec[3 * e + 1], zc = vec[3 * e + 2];
        float inv = rsqrtf(x * x + y * y + zc * zc);
        unit[3 * e]     = x * inv;
        unit[3 * e + 1] = y * inv;
        unit[3 * e + 2] = zc * inv;
    }
}

__global__ void apply_acc_kernel(const float* __restrict__ acc, float* __restrict__ s,
                                 float* __restrict__ v)
{
    const float inv_sqrtH = 0.08838834764831845f; // 1/sqrt(128)
    long idx = (long)blockIdx.x * blockDim.x + threadIdx.x;
    if (idx < (long)N_NODES * 512) {
        int n = (int)(idx >> 9);
        int j = (int)(idx & 511);
        if (j < 128) s[(long)n * 128 + j] += acc[idx];
        else         v[(long)n * 384 + (j - 128)] += acc[idx] * inv_sqrtH;
    }
}

__global__ void feat_kernel(const float* __restrict__ vv, const float* __restrict__ s,
                            float* __restrict__ feat, float* __restrict__ vdot)
{
    const float inv_sqrtH = 0.08838834764831845f;
    long idx = (long)blockIdx.x * blockDim.x + threadIdx.x;
    if (idx < (long)N_NODES * 128) {
        int n = (int)(idx >> 7);
        int c = (int)(idx & 127);
        const float* base = vv + (long)n * 768;
        float dsum = 0.0f, nsum = 0.0f;
        #pragma unroll
        for (int d = 0; d < 3; d++) {
            float a = base[d * 256 + c];
            float b = base[d * 256 + 128 + c];
            dsum += a * b;
            nsum += b * b;
        }
        vdot[idx] = dsum * inv_sqrtH;
        feat[(long)n * 256 + c]       = s[idx];
        feat[(long)n * 256 + 128 + c] = sqrtf(nsum + 1e-12f);
    }
}

__global__ void update2_kernel(const float* __restrict__ uf, const float* __restrict__ vdot,
                               const float* __restrict__ vv, float* __restrict__ s,
                               float* __restrict__ v)
{
    const float inv_sqrt2 = 0.7071067811865476f;
    long idx = (long)blockIdx.x * blockDim.x + threadIdx.x;
    if (idx < (long)N_NODES * 128) {
        int n = (int)(idx >> 7);
        int c = (int)(idx & 127);
        const float* ufr = uf + (long)n * 384;
        float a_ss = ufr[c], a_sv = ufr[128 + c], a_v = ufr[256 + c];
        s[idx] += (vdot[idx] * a_sv + a_ss) * inv_sqrt2;
        const float* base = vv + (long)n * 768;
        #pragma unroll
        for (int d = 0; d < 3; d++)
            v[(long)n * 384 + d * 128 + c] += base[d * 256 + c] * a_v;
    }
}

__global__ void final_kernel(const float* __restrict__ s, const float* __restrict__ w1,
                             const float* __restrict__ b1, const float* __restrict__ w2,
                             const float* __restrict__ b2, float* __restrict__ red)
{
    int wave = threadIdx.x >> 5;
    int lane = threadIdx.x & 31;
    for (long n = (long)blockIdx.x * 8 + wave; n < N_NODES; n += (long)gridDim.x * 8) {
        const float* sr = s + n * HDIM;
        int c0 = lane, c1 = lane + 32;
        float d0 = 0.0f, d1 = 0.0f;
        for (int k = 0; k < HDIM; k++) {
            float sv = sr[k];
            d0 += sv * w1[k * 64 + c0];
            d1 += sv * w1[k * 64 + c1];
        }
        float val = silu_f(d0 + b1[c0]) * w2[c0] + silu_f(d1 + b1[c1]) * w2[c1];
        #pragma unroll
        for (int off = 16; off > 0; off >>= 1) val += __shfl_down(val, off, 32);
        if (lane == 0) atomAddF(red, val + b2[0]);
    }
}

__global__ void finalize_kernel(const float* __restrict__ red, float* __restrict__ out)
{
    out[0] = red[0] / (float)N_NODES;
}

// ---------------------------------------------------------------------------
// Host side
// ---------------------------------------------------------------------------
static inline void launch_gemm(const float* A, const float* B, const float* bias, float* C,
                               int M, int K, int Ncols, bool act, hipStream_t st)
{
    int blocks = ((M + 63) / 64) * (Ncols / 64);
    if (K == 128)
        gemm_tiled<128><<<blocks, 256, 0, st>>>(A, B, bias, C, M, Ncols, act ? 1 : 0);
    else
        gemm_tiled<256><<<blocks, 256, 0, st>>>(A, B, bias, C, M, Ncols, act ? 1 : 0);
}

extern "C" void kernel_launch(void* const* d_in, const int* in_sizes, int n_in,
                              void* d_out, int out_size, void* d_ws, size_t ws_size,
                              hipStream_t stream)
{
    const int*   z      = (const int*)  d_in[0];
    const int*   eidx   = (const int*)  d_in[1];
    const float* edist  = (const float*)d_in[2];
    const float* evec   = (const float*)d_in[3];
    const float* emb    = (const float*)d_in[4];
    const float* msg_w1 = (const float*)d_in[5];
    const float* msg_b1 = (const float*)d_in[6];
    const float* msg_w2 = (const float*)d_in[7];
    const float* msg_b2 = (const float*)d_in[8];
    const float* rbf_w  = (const float*)d_in[9];
    const float* rbf_b  = (const float*)d_in[10];
    const float* upd_vw = (const float*)d_in[11];
    const float* upd_w1 = (const float*)d_in[12];
    const float* upd_b1 = (const float*)d_in[13];
    const float* upd_w2 = (const float*)d_in[14];
    const float* upd_b2 = (const float*)d_in[15];
    const float* out_w1 = (const float*)d_in[16];
    const float* out_b1 = (const float*)d_in[17];
    const float* out_w2 = (const float*)d_in[18];
    const float* out_b2 = (const float*)d_in[19];
    float* out = (float*)d_out;

    const long Ns = (long)N_NODES;
    float* ws   = (float*)d_ws;
    float* s    = ws;                          // N*128
    float* v    = s    + Ns * 128;             // N*384
    float* unit = v    + Ns * 384;             // E*3
    float* t    = unit + (long)N_EDGES * 3;    // N*128
    float* mf   = t    + Ns * 128;             // N*384 (later: feat N*256 + vdot N*128)
    float* accb = mf   + Ns * 384;             // N*512 (later: uf N*384)
    float* vv   = accb + Ns * 512;             // N*768
    float* red  = vv   + Ns * 768;             // 1
    float* feat = mf;
    float* vdot = mf + Ns * 256;
    float* uf   = accb;

    {
        long tot = Ns * 128;
        embed_kernel<<<(int)((tot + 255) / 256), 256, 0, stream>>>(z, emb, s);
    }
    hipMemsetAsync(v, 0, Ns * 384 * sizeof(float), stream);
    unit_kernel<<<(N_EDGES + 255) / 256, 256, 0, stream>>>(evec, unit);

    for (int l = 0; l < LAYERS; l++) {
        launch_gemm(s, msg_w1 + (long)l * 128 * 128, msg_b1 + l * 128, t,
                    N_NODES, 128, 128, true, stream);
        launch_gemm(t, msg_w2 + (long)l * 128 * 384, msg_b2 + l * 384, mf,
                    N_NODES, 128, 384, false, stream);

        hipMemsetAsync(accb, 0, Ns * 512 * sizeof(float), stream);
        edge_kernel<<<4096, 256, 0, stream>>>(eidx, edist, unit,
                                              rbf_w + (long)l * RDIM * 384,
                                              rbf_b + l * 384, mf, v, accb);
        {
            long tot = Ns * 512;
            apply_acc_kernel<<<(int)((tot + 255) / 256), 256, 0, stream>>>(accb, s, v);
        }

        launch_gemm(v, upd_vw + (long)l * 128 * 256, nullptr, vv,
                    3 * N_NODES, 128, 256, false, stream);
        {
            long tot = Ns * 128;
            feat_kernel<<<(int)((tot + 255) / 256), 256, 0, stream>>>(vv, s, feat, vdot);
        }
        launch_gemm(feat, upd_w1 + (long)l * 256 * 128, upd_b1 + l * 128, t,
                    N_NODES, 256, 128, true, stream);
        launch_gemm(t, upd_w2 + (long)l * 128 * 384, upd_b2 + l * 384, uf,
                    N_NODES, 128, 384, false, stream);
        {
            long tot = Ns * 128;
            update2_kernel<<<(int)((tot + 255) / 256), 256, 0, stream>>>(uf, vdot, vv, s, v);
        }
    }

    hipMemsetAsync(red, 0, sizeof(float), stream);
    final_kernel<<<1024, 256, 0, stream>>>(s, out_w1, out_b1, out_w2, out_b2, red);
    finalize_kernel<<<1, 1, 0, stream>>>(red, out);
}
